// SNN_82789789597746
// MI455X (gfx1250) — matrix operand order repaired and checked by reading, not compiled, not run on hardware
//
#include <hip/hip_runtime.h>

typedef __attribute__((ext_vector_type(16))) _Float16 v16h;
typedef __attribute__((ext_vector_type(8)))  _Float16 v8h;
typedef __attribute__((ext_vector_type(4)))  _Float16 v4h;
typedef __attribute__((ext_vector_type(8)))  float    v8f;
typedef __attribute__((ext_vector_type(4)))  float    v4f;

#define T_STEPS 100
#define BATCH   1024
#define D_IN    784
#define D_INP   800      // K of GEMM0 padded to multiple of 32
#define D_H     1024
#define D_OUT   26
#define D_OUTP  32       // N of GEMM1 padded to 2 WMMA tiles
#define BT      16       // batch rows per workgroup (one WMMA M tile)
#define NTHR    512      // 16 waves
#define NT0     4        // GEMM0 N-tiles per wave (16 waves * 64 cols = 1024)
#define THRV    1.0f
#define BETAV   0.8f

// ---------------------------------------------------------------------------
// Prep: W0 [1024,784] f32 -> W0h [1024,800] f16 (K zero-padded)
//       W1 [26,1024]  f32 -> W1h [32,1024]  f16 (N zero-padded)
// ---------------------------------------------------------------------------
__global__ __launch_bounds__(256) void snn_prep(const float* __restrict__ W0,
                                                const float* __restrict__ W1,
                                                _Float16* __restrict__ W0h,
                                                _Float16* __restrict__ W1h) {
  int idx    = blockIdx.x * blockDim.x + threadIdx.x;
  int stride = gridDim.x * blockDim.x;
  const int total0 = D_H * D_INP;
  const int total1 = D_OUTP * D_H;
  for (int i = idx; i < total0; i += stride) {
    int n = i / D_INP, k = i - n * D_INP;
    W0h[i] = (k < D_IN) ? (_Float16)W0[n * D_IN + k] : (_Float16)0.f;
  }
  for (int i = idx; i < total1; i += stride) {
    int n = i / D_H, k = i - n * D_H;
    W1h[i] = (n < D_OUT) ? (_Float16)W1[n * D_H + k] : (_Float16)0.f;
  }
}

// ---------------------------------------------------------------------------
// Fused LIF scan. One workgroup = 16 batch rows, full T loop, state resident.
// 16 waves: each owns 64 hidden cols (4 WMMA N-tiles, 32 VGPRs of membrane).
// ---------------------------------------------------------------------------
__global__ __launch_bounds__(NTHR) void snn_scan(const float*    __restrict__ x,
                                                 const _Float16* __restrict__ W0h,
                                                 const float*    __restrict__ b0,
                                                 const _Float16* __restrict__ W1h,
                                                 const float*    __restrict__ b1,
                                                 float*          __restrict__ out) {
  __shared__ __align__(16) _Float16 Ax[BT * D_INP];     // 25600 B
  __shared__ __align__(16) _Float16 S0[BT * D_H];       // 32768 B
  __shared__ __align__(16) float    CUR1[BT * D_OUTP];  //  2048 B
  __shared__ __align__(16) float    M1[BT * D_OUTP];    //  2048 B

  const int tid    = threadIdx.x;
  const int wave   = tid >> 5;           // 0..15
  const int lane   = tid & 31;
  const int nloc   = lane & 15;
  const int hi     = lane >> 4;          // lane group: 0 = lanes 0-15
  const int aoff   = hi * 8;             // A frag per-lane K-half offset
  const int bkoff  = hi * 8;             // B frag per-lane K-half offset
  const int batch0 = blockIdx.x * BT;
  const int nbase  = wave * (NT0 * 16);  // wave's N base in D_H (64 cols)

  // Single base pointers; all tile offsets become constant immediates.
  const _Float16* bbase0 = W0h + (size_t)(nbase + nloc) * D_INP + bkoff;
  const _Float16* bbase1 = W1h + (size_t)nloc * D_H + wave * 64 + bkoff;
  const _Float16* abase0 = &Ax[nloc * D_INP + aoff];
  const _Float16* abase1 = &S0[nloc * D_H + wave * 64 + aoff];

  float bias0[NT0];
  #pragma unroll
  for (int nt = 0; nt < NT0; ++nt) bias0[nt] = b0[nbase + nt * 16 + nloc];

  // Hidden membrane state == WMMA accumulators: 4 N-tiles x v8f per lane.
  v8f m0s[NT0] = {};

  for (int e = tid; e < BT * D_OUTP; e += NTHR) M1[e] = 0.f;

  const size_t xstep  = (size_t)BATCH * D_IN;
  const size_t outOff = (size_t)T_STEPS * BATCH * D_OUT;

  for (int t = 0; t < T_STEPS; ++t) {
    __syncthreads();  // protect LDS reuse from previous iteration

    // ---- stage x_t tile (f32 -> f16, NT loads), zero pad, CUR1 = b1 ----
    {
      const v4f* xt4 =
          (const v4f*)(x + (size_t)t * xstep + (size_t)batch0 * D_IN);
      for (int i = tid; i < BT * (D_IN / 4); i += NTHR) {
        int r = i / (D_IN / 4), c4 = i - r * (D_IN / 4);
        v4f v = __builtin_nontemporal_load(&xt4[(size_t)r * (D_IN / 4) + c4]);
        v4h h;
        h[0] = (_Float16)v[0]; h[1] = (_Float16)v[1];
        h[2] = (_Float16)v[2]; h[3] = (_Float16)v[3];
        *(v4h*)&Ax[r * D_INP + c4 * 4] = h;
      }
      if (tid < BT * 4) {  // zero K pad: 16 rows x 16 halfs
        int r = tid >> 2, k4 = tid & 3;
        *(v4h*)&Ax[r * D_INP + D_IN + k4 * 4] = (v4h){};
      }
      for (int e = tid; e < BT * D_OUTP; e += NTHR) {
        int n = e & 31;
        CUR1[e] = (n < D_OUT) ? b1[n] : 0.f;
      }
      if (t + 1 < T_STEPS) {  // prefetch next timestep's x tile
        const float* xn = x + (size_t)(t + 1) * xstep + (size_t)batch0 * D_IN;
        for (int i = tid * 32; i < BT * D_IN; i += NTHR * 32)
          __builtin_prefetch(xn + i, 0, 0);
      }
    }

    // ---- in-place LIF pre-transform: m <- beta*m - reset*thr + b0 ----
    #pragma unroll
    for (int nt = 0; nt < NT0; ++nt) {
      float bv = bias0[nt];
      #pragma unroll
      for (int j = 0; j < 8; ++j) {
        float m = m0s[nt][j];
        float reset = (m > THRV) ? THRV : 0.f;
        m0s[nt][j] = BETAV * m - reset + bv;
      }
    }
    __syncthreads();

    // ---- GEMM0: m0s += x_t[16,800] @ W0h^T tile ----
    for (int ks = 0; ks < D_INP / 32; ++ks) {
      v8h alo = *(const v8h*)(abase0 + ks * 32);
      v8h ahi = *(const v8h*)(abase0 + ks * 32 + 16);
      v16h a = __builtin_shufflevector(alo, ahi, 0,1,2,3,4,5,6,7,
                                                 8,9,10,11,12,13,14,15);
      #pragma unroll
      for (int nt = 0; nt < NT0; ++nt) {
        const _Float16* bp = bbase0 + nt * (16 * D_INP) + ks * 32;
        v8h blo = *(const v8h*)(bp);
        v8h bhi = *(const v8h*)(bp + 16);
        v16h b = __builtin_shufflevector(blo, bhi, 0,1,2,3,4,5,6,7,
                                                   8,9,10,11,12,13,14,15);
        m0s[nt] = __builtin_amdgcn_wmma_f32_16x16x32_f16(
            false, a, false, b, (short)0, m0s[nt], false, false);
      }
    }

    // ---- hidden spikes -> LDS (f16) ----
    #pragma unroll
    for (int nt = 0; nt < NT0; ++nt) {
      int ncol = nbase + nt * 16 + nloc;
      #pragma unroll
      for (int j = 0; j < 8; ++j) {
        int row = hi * 8 + j;  // C layout: VGPR j holds M = j + 8*hi
        S0[row * D_H + ncol] =
            (m0s[nt][j] > THRV) ? (_Float16)1.f : (_Float16)0.f;
      }
    }
    __syncthreads();

    // ---- GEMM1: cur1[16,32] = s0 @ W1h^T, K=1024 split across 16 waves ----
    {
      v8f acc1[2] = {};
      #pragma unroll
      for (int ks = 0; ks < 2; ++ks) {
        v8h alo = *(const v8h*)(abase1 + ks * 32);
        v8h ahi = *(const v8h*)(abase1 + ks * 32 + 16);
        v16h a = __builtin_shufflevector(alo, ahi, 0,1,2,3,4,5,6,7,
                                                   8,9,10,11,12,13,14,15);
        #pragma unroll
        for (int nt = 0; nt < 2; ++nt) {
          const _Float16* bp = bbase1 + nt * (16 * D_H) + ks * 32;
          v8h blo = *(const v8h*)(bp);
          v8h bhi = *(const v8h*)(bp + 16);
          v16h b = __builtin_shufflevector(blo, bhi, 0,1,2,3,4,5,6,7,
                                                     8,9,10,11,12,13,14,15);
          acc1[nt] = __builtin_amdgcn_wmma_f32_16x16x32_f16(
              false, a, false, b, (short)0, acc1[nt], false, false);
        }
      }
      #pragma unroll
      for (int nt = 0; nt < 2; ++nt) {
        int n = nt * 16 + nloc;
        #pragma unroll
        for (int j = 0; j < 8; ++j) {
          int row = hi * 8 + j;
          atomicAdd(&CUR1[row * D_OUTP + n], acc1[nt][j]);  // ds_add_f32
        }
      }
    }
    __syncthreads();

    // ---- output LIF update + NT store of spk_rec / mem_rec ----
    for (int e = tid; e < BT * D_OUTP; e += NTHR) {
      int row = e >> 5, n = e & 31;
      float m = M1[e];
      float reset = (m > THRV) ? THRV : 0.f;
      m = BETAV * m + CUR1[e] - reset;
      M1[e] = m;
      if (n < D_OUT) {
        size_t o = ((size_t)t * BATCH + batch0 + row) * D_OUT + n;
        __builtin_nontemporal_store((m > THRV) ? 1.f : 0.f, &out[o]);
        __builtin_nontemporal_store(m, &out[outOff + o]);
      }
    }
  }
}

extern "C" void kernel_launch(void* const* d_in, const int* in_sizes, int n_in,
                              void* d_out, int out_size, void* d_ws, size_t ws_size,
                              hipStream_t stream) {
  const float* x  = (const float*)d_in[0];  // [100,1024,784]
  const float* W0 = (const float*)d_in[1];  // [1024,784]
  const float* b0 = (const float*)d_in[2];  // [1024]
  const float* W1 = (const float*)d_in[3];  // [26,1024]
  const float* b1 = (const float*)d_in[4];  // [26]
  float* out = (float*)d_out;               // spk_rec ++ mem_rec

  _Float16* W0h = (_Float16*)d_ws;                 // 1024*800 f16
  _Float16* W1h = W0h + (size_t)D_H * D_INP;       // 32*1024 f16

  snn_prep<<<416, 256, 0, stream>>>(W0, W1, W0h, W1h);
  snn_scan<<<BATCH / BT, NTHR, 0, stream>>>(x, W0h, b0, W1h, b1, out);
}
